// RNN_MultiRegional_6382321401933
// MI455X (gfx1250) — matrix operand order repaired and checked by reading, not compiled, not run on hardware
//
#include <hip/hip_runtime.h>
#include <hip/hip_bf16.h>
#include <math.h>

#define HID      1024
#define HALF     512
#define N_EXC    359          // HALF - int(0.3*HALF) = 512 - 153
#define BATCH    64
#define TSTEPS   500
#define NBLK     64           // persistent scan blocks (one j-tile each)
#define LDS_PITCH 2064        // 1024 bf16 (2048B) + 16B pad -> conflict-free b128

typedef __attribute__((ext_vector_type(16))) __bf16 bf16x16;
typedef __attribute__((ext_vector_type(8)))  float  f32x8;

union Frag { bf16x16 v; uint4 q[2]; };

// workspace layout (bytes)
#define WS_SYNC   0
#define WS_WBF16  256                                   // 1024*1024 bf16 = 2 MB
#define WS_HBF16  (WS_WBF16 + HID*HID*2)                // 2 x 64*1024 bf16
#define WS_HF32   (WS_HBF16 + 2*BATCH*HID*2)            // 64*1024 f32

__device__ __forceinline__ float relu_f(float x) { return x > 0.f ? x : 0.f; }

// ---------------------------------------------------------------------------
// Build W_rec (bf16, row-major [j][k]) + init h state + zero barrier counters
// ---------------------------------------------------------------------------
__global__ void k_setup(const float* __restrict__ hn,
                        const float* __restrict__ s2s_fixed,
                        const float* __restrict__ s2a_w,
                        const float* __restrict__ a2a_w,
                        const float* __restrict__ a2s_w,
                        unsigned char* __restrict__ ws)
{
    __bf16* Wb  = (__bf16*)(ws + WS_WBF16);
    __bf16* hb0 = (__bf16*)(ws + WS_HBF16);
    float*  hf  = (float*)(ws + WS_HF32);
    unsigned* sync = (unsigned*)(ws + WS_SYNC);

    int tid = blockIdx.x * blockDim.x + threadIdx.x;
    int stride = gridDim.x * blockDim.x;
    if (tid == 0) { sync[0] = 0u; sync[1] = 0u; }

    for (int i = tid; i < HID * HID; i += stride) {
        int j = i >> 10, k = i & (HID - 1);
        float v;
        if (j < HALF) {
            if (k < HALF) {                       // str2str: fixed @ (-I)
                v = -s2s_fixed[j * HALF + k];
            } else {                              // alm2str: mask * relu
                int k2 = k - HALF;
                v = (k2 < N_EXC) ? relu_f(a2s_w[j * HALF + k2]) : 0.f;
            }
        } else {
            int j2 = j - HALF;
            if (k < HALF) {                       // str2alm: relu
                v = relu_f(s2a_w[j2 * HALF + k]);
            } else {                              // alm2alm: relu @ diag(+/-1)
                int k2 = k - HALF;
                float r = relu_f(a2a_w[j2 * HALF + k2]);
                v = (k2 < N_EXC) ? r : -r;
            }
        }
        Wb[i] = (__bf16)v;
    }
    for (int i = tid; i < BATCH * HID; i += stride) {
        float h = hn[i];
        hf[i] = h;
        hb0[i] = (__bf16)h;
    }
}

// ---------------------------------------------------------------------------
// Persistent recurrent scan: 64 blocks x 4 waves = 256 WMMA tiles of h@W^T.
// Block owns 16 output columns (W rows cached in LDS); wave owns 16 batch rows.
// Device-wide software barrier between timesteps.
// ---------------------------------------------------------------------------
__global__ void __launch_bounds__(128) k_scan(const float* __restrict__ inp,
                                              const float* __restrict__ inp_w,
                                              float* __restrict__ rnn_out,
                                              unsigned char* __restrict__ ws)
{
    __shared__ __align__(16) unsigned char smem[16 * LDS_PITCH];

    const __bf16* Wb = (const __bf16*)(ws + WS_WBF16);
    __bf16* hbuf = (__bf16*)(ws + WS_HBF16);
    float*  hf   = (float*)(ws + WS_HF32);
    volatile unsigned* scount = (volatile unsigned*)(ws + WS_SYNC);
    volatile unsigned* sgen   = scount + 1;

    const int wave  = threadIdx.x >> 5;
    const int lane  = threadIdx.x & 31;
    const int ntile = blockIdx.x;      // j-tile (0..63)
    const int mtile = wave;            // b-tile (0..3)
    const int m  = lane & 15;
    const int hi = lane >> 4;

    // Stage 16 rows of W (this block's output columns) into padded LDS.
    {
        const uint4* src = (const uint4*)(Wb + (size_t)ntile * 16 * HID);
        for (int it = 0; it < 16; ++it) {
            int idx = it * 128 + threadIdx.x;     // 2048 uint4s total
            int r = idx >> 7, c = idx & 127;
            *(uint4*)(smem + r * LDS_PITCH + c * 16) = src[r * 128 + c];
        }
    }
    __syncthreads();

    const int j = ntile * 16 + m;                 // this lane's output column
    const float tc = (j < HALF) ? 0.002f : 0.1f;
    float w0 = 0.f, w1 = 0.f, w2 = 0.f, w3 = 0.f; // str_mask folds into u_t
    if (j < HALF) {
        w0 = inp_w[0 * HID + j]; w1 = inp_w[1 * HID + j];
        w2 = inp_w[2 * HID + j]; w3 = inp_w[3 * HID + j];
    }

    for (int t = 0; t < TSTEPS; ++t) {
        const __bf16* hcur  = hbuf + (size_t)(t & 1) * (BATCH * HID);
        __bf16*       hnext = hbuf + (size_t)((t + 1) & 1) * (BATCH * HID);
        const __bf16* hrow  = hcur + (size_t)(mtile * 16 + m) * HID;

        f32x8 acc = {0.f, 0.f, 0.f, 0.f, 0.f, 0.f, 0.f, 0.f};
        #pragma unroll
        for (int c = 0; c < 32; ++c) {
            Frag af, bf;
            // A (16x32 bf16): lane half-dependent K runs {0..7,16..23}(+8)
            af.q[0] = *(const uint4*)(hrow + c * 32 + hi * 8);
            af.q[1] = *(const uint4*)(hrow + c * 32 + 16 + hi * 8);
            // B (32x16 bf16): lane n=m holds W row j, K runs {0..7,16..23}(+8)
            const unsigned char* bp = smem + m * LDS_PITCH + (c * 32 + hi * 8) * 2;
            bf.q[0] = *(const uint4*)(bp);
            bf.q[1] = *(const uint4*)(bp + 32);
            acc = __builtin_amdgcn_wmma_f32_16x16x32_bf16(
                      false, af.v, false, bf.v, (short)0, acc, false, false);
        }

        // Epilogue: h_new = relu((1-tc)*h + tc*(acc + u_t)), fused u_t
        #pragma unroll
        for (int r = 0; r < 8; ++r) {
            int bb = mtile * 16 + hi * 8 + r;     // batch row (D layout)
            float4 iv = *(const float4*)(inp + ((size_t)bb * TSTEPS + t) * 4);
            float u = iv.x * w0 + iv.y * w1 + iv.z * w2 + iv.w * w3;
            float hold = hf[bb * HID + j];
            float hnew = (1.f - tc) * hold + tc * (acc[r] + u);
            hnew = hnew > 0.f ? hnew : 0.f;
            hf[bb * HID + j] = hnew;
            rnn_out[((size_t)bb * TSTEPS + t) * HID + j] = hnew;
            hnext[bb * HID + j] = (__bf16)hnew;
        }

        // device-wide barrier between timesteps (sense via generation count)
        __threadfence();
        __syncthreads();
        if (threadIdx.x == 0) {
            unsigned g = *sgen;
            if (atomicAdd((unsigned*)scount, 1u) == NBLK - 1) {
                *scount = 0u;
                __threadfence();
                atomicAdd((unsigned*)sgen, 1u);
            } else {
                while (*sgen == g) __builtin_amdgcn_s_sleep(1);
            }
            __threadfence();
        }
        __syncthreads();
    }
}

// ---------------------------------------------------------------------------
// out[b,t] = sigmoid(dot(rnn[b,t,512:], fc1_w[512:]) + fc1_b)  (wave32 dot)
// ---------------------------------------------------------------------------
__global__ void k_head(const float* __restrict__ rnn,
                       const float* __restrict__ fc1_w,
                       const float* __restrict__ fc1_b,
                       float* __restrict__ outp)
{
    int wid = blockIdx.x * (blockDim.x >> 5) + (threadIdx.x >> 5);
    int lane = threadIdx.x & 31;
    if (wid >= BATCH * TSTEPS) return;
    int b = wid / TSTEPS, t = wid % TSTEPS;
    const float* row = rnn + ((size_t)b * TSTEPS + t) * HID;
    float s = 0.f;
    for (int h = HALF + lane; h < HID; h += 32) s += row[h] * fc1_w[h];
    #pragma unroll
    for (int off = 16; off > 0; off >>= 1) s += __shfl_xor(s, off, 32);
    if (lane == 0) outp[b * TSTEPS + t] = 1.f / (1.f + __expf(-(s + fc1_b[0])));
}

__global__ void k_last(const float* __restrict__ rnn, const float* __restrict__ x_in,
                       float* __restrict__ hn_last, float* __restrict__ x_last)
{
    int i = blockIdx.x * blockDim.x + threadIdx.x;
    int stride = gridDim.x * blockDim.x;
    for (; i < BATCH * HID; i += stride) {
        int b = i >> 10, jj = i & (HID - 1);
        hn_last[i] = rnn[((size_t)b * TSTEPS + (TSTEPS - 1)) * HID + jj];
        x_last[i]  = x_in[i];
    }
}

__global__ void k_xout(const float* __restrict__ x_in, float* __restrict__ x_out)
{
    size_t i = (size_t)blockIdx.x * blockDim.x + threadIdx.x;
    size_t stride = (size_t)gridDim.x * blockDim.x;
    size_t total = (size_t)BATCH * TSTEPS * HID;
    for (; i < total; i += stride) {
        size_t b = i / ((size_t)TSTEPS * HID);
        size_t jj = i & (HID - 1);
        x_out[i] = x_in[b * HID + jj];
    }
}

// ---------------------------------------------------------------------------
extern "C" void kernel_launch(void* const* d_in, const int* in_sizes, int n_in,
                              void* d_out, int out_size, void* d_ws, size_t ws_size,
                              hipStream_t stream)
{
    (void)in_sizes; (void)n_in; (void)out_size; (void)ws_size;
    const float* inp       = (const float*)d_in[0];
    const float* hn        = (const float*)d_in[1];
    const float* x_in      = (const float*)d_in[2];
    /* d_in[3] str2str_w: multiplied by 0 in the reference -> unused */
    const float* s2a_w     = (const float*)d_in[4];
    const float* a2a_w     = (const float*)d_in[5];
    const float* a2s_w     = (const float*)d_in[6];
    const float* s2s_fixed = (const float*)d_in[7];
    const float* inp_w     = (const float*)d_in[8];
    const float* fc1_w     = (const float*)d_in[9];
    const float* fc1_b     = (const float*)d_in[10];

    float* out     = (float*)d_out;                          // [B,T,1]
    float* hn_last = out + BATCH * TSTEPS;                   // [1,B,HID]
    float* rnn     = hn_last + BATCH * HID;                  // [B,T,HID]
    float* x_last  = rnn + (size_t)BATCH * TSTEPS * HID;     // [1,B,HID]
    float* x_out   = x_last + BATCH * HID;                   // [B,T,HID]
    unsigned char* ws = (unsigned char*)d_ws;

    k_setup<<<256, 256, 0, stream>>>(hn, s2s_fixed, s2a_w, a2a_w, a2s_w, ws);
    k_scan <<<NBLK, 128, 0, stream>>>(inp, inp_w, rnn, ws);
    k_head <<<(BATCH * TSTEPS) / 8, 256, 0, stream>>>(rnn, fc1_w, fc1_b, out);
    k_last <<<64, 256, 0, stream>>>(rnn, x_in, hn_last, x_last);
    k_xout <<<2048, 256, 0, stream>>>(x_in, x_out);
}
